// FFEAdapter_3453153706603
// MI455X (gfx1250) — matrix operand order repaired and checked by reading, not compiled, not run on hardware
//
#include <hip/hip_runtime.h>
#include <hip/hip_bf16.h>

// ---------------- types ----------------
typedef __attribute__((ext_vector_type(16))) __bf16 v16bf;
typedef __attribute__((ext_vector_type(8)))  float  v8f;
typedef __attribute__((ext_vector_type(4)))  unsigned int v4u;
typedef __attribute__((address_space(3))) unsigned short lds_ushort;

union FragBf16 { v4u u[2]; v16bf v; };

// ---------------- constants ----------------
#define BB   16
#define HH   64
#define WW   64
#define DD   768
#define HB   22            // ceil(64/3)
#define WB   22
#define NWIN (BB*HB*WB)    // 7744
#define MROWS (NWIN*9)     // 69696
#define MTILE 144          // 16 windows * 9 rows
#define NTILE 128
#define KC    32
#define KT    (DD/KC)      // 24
#define NWAVE 9            // 288 threads
#define NTHR  (NWAVE*32)

// staging buffer layout (ushort units): Ahi 4608 | Alo 4608 | Bhi 4096 | Blo 4096
#define BUF_USH   17408    // 34816 bytes per stage buffer
#define SMEM_BYTES (2*BUF_USH*2)   // 69632 B (sY 36864B aliases this)

__device__ __constant__ float Cd[3][3] = {
  { 0.57735026918962576f,  0.57735026918962576f,  0.57735026918962576f},
  { 0.70710678118654752f,  0.0f,                 -0.70710678118654752f},
  { 0.40824829046386302f, -0.81649658092772603f,  0.40824829046386302f}};

// ---------------- bf16 helpers (RNE) ----------------
__device__ __forceinline__ unsigned short f2bf(float x) {
  unsigned u = __builtin_bit_cast(unsigned, x);
  unsigned r = (u + 0x7FFFu + ((u >> 16) & 1u)) >> 16;
  return (unsigned short)r;
}
__device__ __forceinline__ float bf2f(unsigned short h) {
  unsigned u = ((unsigned)h) << 16;
  return __builtin_bit_cast(float, u);
}

// ---------------- CDNA5 async copy primitives ----------------
__device__ __forceinline__ void async_load_b128(const void* gsrc, unsigned ldsByteOff) {
  // GLOBAL_LOAD_ASYNC_TO_LDS_B128: vdst = 32-bit LDS byte offset, vaddr = 64-bit global addr
  asm volatile("global_load_async_to_lds_b128 %0, %1, off"
               :: "v"(ldsByteOff), "v"(gsrc) : "memory");
}
__device__ __forceinline__ void wait_asynccnt0() {
#if __has_builtin(__builtin_amdgcn_s_wait_asynccnt)
  __builtin_amdgcn_s_wait_asynccnt(0);
#else
  asm volatile("s_wait_asynccnt 0x0" ::: "memory");
#endif
}

// ============================================================
// Kernel 1: edge-pad + 2D 3x3 DCT, emit A as hi/lo bf16 split.
// ============================================================
__global__ void ffe_prep_dct(const float* __restrict__ f,
                             unsigned short* __restrict__ aHi,
                             unsigned short* __restrict__ aLo) {
  int idx = blockIdx.x * 256 + threadIdx.x;   // NWIN*DD threads exactly
  int d   = idx % DD;
  int win = idx / DD;
  int b  = win / (HB*WB);
  int r  = win - b*(HB*WB);
  int hb = r / WB;
  int wb = r - hb*WB;

  float v[3][3];
#pragma unroll
  for (int i = 0; i < 3; ++i) {
    int h = hb*3 + i; if (h > HH-1) h = HH-1;   // edge pad
#pragma unroll
    for (int j = 0; j < 3; ++j) {
      int w = wb*3 + j; if (w > WW-1) w = WW-1;
      v[i][j] = f[(((size_t)b*HH + h)*WW + w)*DD + d];
    }
  }
  float tmp[3][3];
#pragma unroll
  for (int p = 0; p < 3; ++p)
#pragma unroll
    for (int j = 0; j < 3; ++j)
      tmp[p][j] = Cd[p][0]*v[0][j] + Cd[p][1]*v[1][j] + Cd[p][2]*v[2][j];

#pragma unroll
  for (int p = 0; p < 3; ++p)
#pragma unroll
    for (int q = 0; q < 3; ++q) {
      float t = tmp[p][0]*Cd[q][0] + tmp[p][1]*Cd[q][1] + tmp[p][2]*Cd[q][2];
      unsigned short hi = f2bf(t);
      unsigned short lo = f2bf(t - bf2f(hi));
      size_t off = ((size_t)(win*9 + p*3 + q))*DD + d;
      aHi[off] = hi;
      aLo[off] = lo;
    }
}

// ============================================================
// Kernel 2: split W1 into hi/lo bf16 (layout stays [o][i] == B^T rows)
// ============================================================
__global__ void ffe_prep_w(const float* __restrict__ W1,
                           unsigned short* __restrict__ wHi,
                           unsigned short* __restrict__ wLo) {
  int idx = blockIdx.x * 256 + threadIdx.x;   // DD*DD threads exactly
  float x = W1[idx];
  unsigned short hi = f2bf(x);
  wHi[idx] = hi;
  wLo[idx] = f2bf(x - bf2f(hi));
}

// ============================================================
// Kernel 3: GEMM (bf16x3) + bias + exact GELU + inverse DCT + crop,
// with double-buffered ASYNC global->LDS staging (ASYNCcnt pipelined).
// grid = (DD/NTILE, MROWS/MTILE) = (6, 484); block = 288 (9 waves)
// ============================================================
__global__ __launch_bounds__(NTHR)
void ffe_gemm_gelu_idct(const unsigned short* __restrict__ aHi,
                        const unsigned short* __restrict__ aLo,
                        const unsigned short* __restrict__ wHi,
                        const unsigned short* __restrict__ wLo,
                        const float* __restrict__ b1,
                        float* __restrict__ out) {
  __shared__ __attribute__((aligned(16))) unsigned char smem[SMEM_BYTES];
  unsigned short* sU = (unsigned short*)smem;
  float* sY = (float*)smem;                      // epilogue reuse (36KB)

  const int tid  = threadIdx.x;
  const int wave = tid >> 5;
  const int lane = tid & 31;
  const int l16  = lane & 15;
  const int kh   = lane >> 4;          // K-half selector

  const int nTileIdx = blockIdx.x;     // fastest-varying -> L2 reuse of A tile
  const int mTileIdx = blockIdx.y;
  const int nBase   = nTileIdx * NTILE;
  const int mBase   = mTileIdx * MTILE;
  const int winBase = mTileIdx * 16;

  // 32-bit LDS byte offset of smem base (addrspace(3) value)
  const unsigned smemLds = (unsigned)(unsigned long long)(lds_ushort*)sU;

  // ---- async stage one K-slice into buffer `buf` (8 b128 ops per thread) ----
  auto stage = [&](int buf, int kBase) {
    const unsigned base = smemLds + (unsigned)buf * (BUF_USH*2);
#pragma unroll
    for (int it = 0; it < 4; ++it) {           // A hi+lo: 1152 chunks exact
      int c  = tid + it*NTHR;
      int hl = c >= 576;
      int cc = c - hl*576;
      int row = cc >> 2, seg = cc & 3;
      const unsigned short* src = (hl ? aLo : aHi) +
          (size_t)(mBase + row)*DD + kBase + seg*8;
      unsigned dst = base + (unsigned)((hl ? 4608 : 0) + row*KC + seg*8) * 2u;
      async_load_b128(src, dst);
    }
#pragma unroll
    for (int it = 0; it < 4; ++it) {           // B hi+lo: 1024 chunks, padded wrap
      int c  = (tid + it*NTHR) & 1023;
      int hl = c >= 512;
      int cc = c - hl*512;
      int row = cc >> 2, seg = cc & 3;
      const unsigned short* src = (hl ? wLo : wHi) +
          (size_t)(nBase + row)*DD + kBase + seg*8;
      unsigned dst = base + (unsigned)(9216 + (hl ? 4096 : 0) + row*KC + seg*8) * 2u;
      async_load_b128(src, dst);
    }
  };

  v8f acc[8];
#pragma unroll
  for (int t = 0; t < 8; ++t) acc[t] = (v8f)0.0f;

  stage(0, 0);                                  // prologue: fill buffer 0

  for (int kt = 0; kt < KT; ++kt) {
    wait_asynccnt0();                           // my writes for buf[kt&1] landed
    __syncthreads();                            // everyone's landed; prev reads done
    if (kt + 1 < KT) stage((kt + 1) & 1, (kt + 1) * KC);  // overlap with compute

    const unsigned short* cur = sU + (kt & 1) * BUF_USH;
    const unsigned short* sAh = cur;
    const unsigned short* sAl = cur + 4608;
    const unsigned short* sBh = cur + 9216;
    const unsigned short* sBl = cur + 13312;

    // A fragments (16-bit A layout: lanes0-15 k0-7/k16-23, lanes16-31 k8-15/k24-31)
    const int rowA = wave*16 + l16;
    FragBf16 fah, fal;
    fah.u[0] = *(const v4u*)(sAh + rowA*KC + kh*8);
    fah.u[1] = *(const v4u*)(sAh + rowA*KC + 16 + kh*8);
    fal.u[0] = *(const v4u*)(sAl + rowA*KC + kh*8);
    fal.u[1] = *(const v4u*)(sAl + rowA*KC + 16 + kh*8);

#pragma unroll
    for (int t = 0; t < 8; ++t) {
      const int nrow = t*16 + l16;
      // B layout: lanes0-15 k0-7/k16-23, lanes16-31 k8-15/k24-31 (matches A)
      FragBf16 fbh, fbl;
      fbh.u[0] = *(const v4u*)(sBh + nrow*KC + kh*8);
      fbh.u[1] = *(const v4u*)(sBh + nrow*KC + 16 + kh*8);
      fbl.u[0] = *(const v4u*)(sBl + nrow*KC + kh*8);
      fbl.u[1] = *(const v4u*)(sBl + nrow*KC + 16 + kh*8);

      acc[t] = __builtin_amdgcn_wmma_f32_16x16x32_bf16(
          false, fah.v, false, fbh.v, (short)0, acc[t], false, false);
      acc[t] = __builtin_amdgcn_wmma_f32_16x16x32_bf16(
          false, fah.v, false, fbl.v, (short)0, acc[t], false, false);
      acc[t] = __builtin_amdgcn_wmma_f32_16x16x32_bf16(
          false, fal.v, false, fbh.v, (short)0, acc[t], false, false);
    }
  }
  __syncthreads();   // all waves done reading staging LDS before sY overwrites it

  // ================= epilogue: bias + GELU -> LDS, IDCT -> out =================
  for (int half = 0; half < 2; ++half) {
#pragma unroll
    for (int tt = 0; tt < 4; ++tt) {
      int t  = half*4 + tt;
      int nl = t*16 + l16;                    // local n in [0,128)
      float bias = b1[nBase + nl];
#pragma unroll
      for (int r = 0; r < 8; ++r) {
        float x = acc[t][r] + bias;
        x = 0.5f * x * (1.0f + erff(x * 0.70710678118654752f));  // exact GELU
        int ml = wave*16 + r + (kh << 3);     // C/D layout: lanes16-31 -> M+8
        sY[ml*64 + (nl - half*64)] = x;
      }
    }
    __syncthreads();

    // inverse DCT over each 3x3 window + crop + store
    for (int e = tid; e < 16*9*64; e += NTHR) {
      int n  = e & 63;
      int pq = (e >> 6) % 9;
      int wl = (e >> 6) / 9;
      int a  = pq / 3, c = pq - a*3;
      float s = 0.0f;
#pragma unroll
      for (int p = 0; p < 3; ++p)
#pragma unroll
        for (int q = 0; q < 3; ++q)
          s += Cd[p][a] * Cd[q][c] * sY[(wl*9 + p*3 + q)*64 + n];

      int wg = winBase + wl;
      int b  = wg / (HB*WB);
      int rr = wg - b*(HB*WB);
      int hb = rr / WB, wb = rr - hb*WB;
      int h = hb*3 + a, w = wb*3 + c;
      if (h < HH && w < WW)
        out[(((size_t)b*HH + h)*WW + w)*DD + nBase + half*64 + n] = s;
    }
    __syncthreads();
  }
}

// ============================================================
extern "C" void kernel_launch(void* const* d_in, const int* in_sizes, int n_in,
                              void* d_out, int out_size, void* d_ws, size_t ws_size,
                              hipStream_t stream) {
  const float* f  = (const float*)d_in[0];
  const float* W1 = (const float*)d_in[1];
  const float* b1 = (const float*)d_in[2];
  float* out = (float*)d_out;

  unsigned short* aHi = (unsigned short*)d_ws;
  unsigned short* aLo = aHi + (size_t)MROWS*DD;
  unsigned short* wHi = aLo + (size_t)MROWS*DD;
  unsigned short* wLo = wHi + (size_t)DD*DD;

  ffe_prep_w<<<(DD*DD)/256, 256, 0, stream>>>(W1, wHi, wLo);
  ffe_prep_dct<<<((size_t)NWIN*DD)/256, 256, 0, stream>>>(f, aHi, aLo);

  dim3 grid(DD/NTILE, MROWS/MTILE);   // (6, 484), N fastest for L2 A-reuse
  ffe_gemm_gelu_idct<<<grid, NTHR, 0, stream>>>(aHi, aLo, wHi, wLo, b1, out);
}
